// FixupBasicBlockT_27986006901520
// MI455X (gfx1250) — matrix operand order repaired and checked by reading, not compiled, not run on hardware
//
#include <hip/hip_runtime.h>

// ---------------- problem constants ----------------
constexpr int BN  = 32;     // batch
constexpr int CP  = 256;    // planes
constexpr int CI  = 128;    // inplanes
constexpr int H   = 28, W = 28;
constexpr int HW  = H * W;            // 784
constexpr int KK  = CP * 9;           // 2304 (GEMM1/2 reduction)
constexpr int N1  = BN * HW;          // 25088
constexpr int OH2 = 55;
constexpr int N2  = BN * OH2 * OH2;   // 96800
constexpr int N3  = BN * HW;          // per-parity output pixels for up branch

// ---------------- workspace layout (bytes) ----------------
constexpr size_t SZ_XBF   = (size_t)BN * CP * HW * 2;   // x+bias1a, bf16
constexpr size_t OFF_XBF  = 0;
constexpr size_t OFF_RELU = OFF_XBF + SZ_XBF;           // relu(conv1)+bias2a, bf16
constexpr size_t OFF_MAIN = OFF_RELU + SZ_XBF;          // main branch f32 [B,CI,55,55]
constexpr size_t SZ_MAIN  = (size_t)BN * CI * OH2 * OH2 * 4;
constexpr size_t OFF_XSEL = OFF_MAIN + SZ_MAIN;         // gathered channels bf16
constexpr size_t SZ_XSEL  = (size_t)BN * CI * HW * 2;
constexpr size_t OFF_W1   = OFF_XSEL + SZ_XSEL;         // [256][2304] bf16
constexpr size_t SZ_W1    = (size_t)CP * KK * 2;
constexpr size_t OFF_W2   = OFF_W1 + SZ_W1;             // [128][2304] bf16
constexpr size_t SZ_W2    = (size_t)CI * KK * 2;
constexpr size_t OFF_W3   = OFF_W2 + SZ_W2;             // [4][128][128] bf16
constexpr size_t SZ_W3    = (size_t)4 * CI * CI * 2;
constexpr size_t OFF_RNG  = OFF_W3 + SZ_W3;             // [B][CP] f32
constexpr size_t OFF_IDX  = OFF_RNG + (size_t)BN * CP * 4;  // [B][CI] i32

// ---------------- WMMA / TDM types ----------------
typedef __bf16 v16bf __attribute__((ext_vector_type(16)));
typedef float  v8f   __attribute__((ext_vector_type(8)));
typedef unsigned int u32x4 __attribute__((ext_vector_type(4)));
typedef int          i32x8 __attribute__((ext_vector_type(8)));
typedef int          i32x4 __attribute__((ext_vector_type(4)));

union FragAB { v16bf v; uint4 q[2]; unsigned short s[16]; };
union AccU   { v8f v; float f[8]; };
union VBuf   { unsigned short s[16]; uint4 q4[2]; };

__device__ __forceinline__ unsigned short f2bf(float f) {
  union { float f; unsigned u; } x; x.f = f;
  unsigned r = x.u + 0x7FFFu + ((x.u >> 16) & 1u);   // round-to-nearest-even
  return (unsigned short)(r >> 16);
}

__device__ __forceinline__ v8f mma_bf16(const FragAB& a, const FragAB& b, v8f c) {
  return __builtin_amdgcn_wmma_f32_16x16x32_bf16(false, a.v, false, b.v,
                                                 (short)0, c, false, false);
}

// A fragment (16x32 bf16): lane L(<16) holds row M=L, K in {0..7,16..23};
// lane L+16 holds row M=L, K in {8..15,24..31}.  LDS row stride = 40 ushorts.
__device__ __forceinline__ void frag_a(const unsigned short* As, int mbase, int lane, FragAB& a) {
  int row = mbase + (lane & 15);
  int kb  = (lane >> 4) << 3;                   // 0 or 8
  a.q[0] = *(const uint4*)(As + row * 40 + kb);
  a.q[1] = *(const uint4*)(As + row * 40 + kb + 16);
}
// B fragment (32x16 bf16): lane L holds col N = L%16, K in {0..7,16..23} (L<16) or {8..15,24..31}.
// Bs stored transposed [n][k], stride 40.
__device__ __forceinline__ void frag_b(const unsigned short* Bs, int nbase, int lane, FragAB& b) {
  int col = nbase + (lane & 15);
  int kb  = (lane >> 4) << 3;                   // 0 or 8
  b.q[0] = *(const uint4*)(Bs + col * 40 + kb);
  b.q[1] = *(const uint4*)(Bs + col * 40 + kb + 16);
}

// -------- Tensor Data Mover: load a dense 64(row) x 32(col) bf16 tile into LDS.
// LDS pad: after every 16 DWORDs (one 32-elem row) insert 4 DWORDs ->
// LDS row stride = 40 ushorts (matches frag_a/frag_b layout).
// D# per CDNA5 ISA (group0: count|lds|global|type=2;
// group1: data_size=2B, pad_enable, interval=16DW, amount=4DW, dims/strides).
// This toolchain's builtin takes 6 args (extra int32x8 before cpol).
__device__ __forceinline__ void tdm_load_a(const unsigned short* src, unsigned ldsOff,
                                           int strideElems) {
  unsigned long long ga = (unsigned long long)(uintptr_t)src;
  u32x4 g0 = { 1u,                                    // count=1, no gather
               ldsOff,                                // lds_addr
               (unsigned)ga,                          // global_addr[31:0]
               (unsigned)((ga >> 32) & 0x1FFFFFFull) | 0x80000000u }; // [56:32] | type=2
  i32x8 g1 = { (1 << 16) | (1 << 20) | (3 << 22) | (3 << 25),
               //  data_size=2B | pad_enable | pad_interval=16DW | pad_amount=4DW
               (32 << 16),          // tensor_dim0 = 32  (bits 63:48)
               (64 << 16),          // tensor_dim1 = 64  (bits 95:80)
               (32 << 16),          // tile_dim0  = 32   (bits 127:112)
               64,                  // tile_dim1  = 64, tile_dim2 = 0
               strideElems,         // tensor_dim0_stride (elements)
               0, 0 };
  i32x4 gz4 = { 0, 0, 0, 0 };
  i32x8 gz8 = { 0, 0, 0, 0, 0, 0, 0, 0 };
  __builtin_amdgcn_tensor_load_to_lds(g0, g1, gz4, gz4, gz8, 0);
}

// ================= pack / prep kernels =================
__global__ void k_pack_x(const float* __restrict__ x, const float* __restrict__ b1a,
                         unsigned short* __restrict__ xbf, int n) {
  int i = blockIdx.x * blockDim.x + threadIdx.x;
  if (i < n) xbf[i] = f2bf(x[i] + b1a[0]);
}
// w_conv2 [ci=256][co=256][3][3] -> wpack1[co][ci*9+q]
__global__ void k_pack_w2(const float* __restrict__ w, unsigned short* __restrict__ wp) {
  int i = blockIdx.x * blockDim.x + threadIdx.x;
  if (i >= CP * KK) return;
  int co = i / KK, k = i - co * KK;
  int ci = k / 9,  q = k - ci * 9;
  wp[i] = f2bf(w[(ci * CP + co) * 9 + q]);
}
// w_conv1 [ci=256][co=128][3][3] -> wpack2[co][ci*9+q]
__global__ void k_pack_w1(const float* __restrict__ w, unsigned short* __restrict__ wp) {
  int i = blockIdx.x * blockDim.x + threadIdx.x;
  if (i >= CI * KK) return;
  int co = i / KK, k = i - co * KK;
  int ci = k / 9,  q = k - ci * 9;
  wp[i] = f2bf(w[(ci * CI + co) * 9 + q]);
}
// w_up [ci=128][co=128][2][2] -> wpack3[p][co][ci],  p = pi*2+pj
__global__ void k_pack_wup(const float* __restrict__ w, unsigned short* __restrict__ wp) {
  int i = blockIdx.x * blockDim.x + threadIdx.x;
  if (i >= 4 * CI * CI) return;
  int p = i / (CI * CI), r = i - p * CI * CI;
  int co = r / CI, ci = r - co * CI;
  wp[i] = f2bf(w[(ci * CI + co) * 4 + p]);
}

// per-(b,c) channel range
__global__ __launch_bounds__(128) void k_range(const float* __restrict__ x, float* __restrict__ rng) {
  __shared__ float smx[128], smn[128];
  int bc = blockIdx.x, t = threadIdx.x;
  const float* p = x + (size_t)bc * HW;
  float mx = -3.4e38f, mn = 3.4e38f;
  for (int i = t; i < HW; i += 128) { float v = p[i]; mx = fmaxf(mx, v); mn = fminf(mn, v); }
  smx[t] = mx; smn[t] = mn; __syncthreads();
  for (int s = 64; s > 0; s >>= 1) {
    if (t < s) { smx[t] = fmaxf(smx[t], smx[t + s]); smn[t] = fminf(smn[t], smn[t + s]); }
    __syncthreads();
  }
  if (t == 0) rng[bc] = smx[0] - smn[0];
}

// per-sample bitonic argsort of 256 ranges; keep top half (ascending order)
__global__ __launch_bounds__(256) void k_argsort(const float* __restrict__ rng, int* __restrict__ idx) {
  __shared__ float key[256]; __shared__ int vid[256];
  int b = blockIdx.x, t = threadIdx.x;
  key[t] = rng[b * CP + t]; vid[t] = t; __syncthreads();
  for (int ks = 2; ks <= 256; ks <<= 1)
    for (int j = ks >> 1; j > 0; j >>= 1) {
      int ixj = t ^ j;
      if (ixj > t) {
        bool up = ((t & ks) == 0);
        float ka = key[t], kb = key[ixj];
        bool sw = up ? (ka > kb) : (ka < kb);
        if (sw) { key[t] = kb; key[ixj] = ka; int tv = vid[t]; vid[t] = vid[ixj]; vid[ixj] = tv; }
      }
      __syncthreads();
    }
  if (t >= 128) idx[b * CI + (t - 128)] = vid[t];
}

// gather selected channels (x already has +bias1a)
__global__ void k_gather(const unsigned short* __restrict__ xbf, const int* __restrict__ idx,
                         unsigned short* __restrict__ xsel) {
  int i = blockIdx.x * blockDim.x + threadIdx.x;
  if (i >= BN * CI * HW) return;
  int pix = i % HW, s = (i / HW) % CI, b = i / (HW * CI);
  xsel[i] = xbf[((size_t)b * CP + idx[b * CI + s]) * HW + pix];
}

// ================= GEMM1: 3x3 s1 p1 conv (256->256), fused relu+bias =================
__global__ __launch_bounds__(256) void k_gemm1(const unsigned short* __restrict__ xbf,
                                               const unsigned short* __restrict__ wp,
                                               unsigned short* __restrict__ relu1,
                                               const float* __restrict__ b1b,
                                               const float* __restrict__ b2a) {
  __shared__ __align__(16) unsigned short As[64 * 40];
  __shared__ __align__(16) unsigned short Bs[128 * 40];
  const int tid = threadIdx.x, lane = tid & 31, wave = tid >> 5;
  const int wm = (wave & 1) << 5, wn = (wave >> 1) << 5;
  const int mBase = blockIdx.y << 6, nBase = blockIdx.x << 7;
  const unsigned AsOff = (unsigned)(uintptr_t)(&As[0]);

  const int nLoc = tid & 127;
  const int kHalf = (tid >> 7) << 4;            // 0 or 16: contiguous 16-k strip
  const int n = nBase + nLoc;
  const int bb = n / HW, rp = n - bb * HW;
  const int oh = rp / W, ow = rp - (rp / W) * W;
  const int ohp = oh + 1, owp = ow + 1;
  const unsigned short* xb = xbf + (size_t)bb * CP * HW;

  v8f acc[2][2] = {};
  for (int kc = 0; kc < KK; kc += 32) {
    if (tid == 0) {                             // TDM DMA of dense A tile
      tdm_load_a(wp + (size_t)mBase * KK + kc, AsOff, KK);
      __builtin_amdgcn_s_wait_tensorcnt(0);
    }
    int kg0 = kc + kHalf;
    int ci = kg0 / 9, q = kg0 - ci * 9;         // once per K-step, not per element
    int kh = q / 3,  kw = q - kh * 3;
    __builtin_prefetch(xb + (ci + 4) * HW + oh * W, 0, 1);
    VBuf vb;
#pragma unroll
    for (int i = 0; i < 16; ++i) {
      int ih = ohp - kh, iw = owp - kw;
      unsigned short v = 0;
      if ((unsigned)ih < 28u && (unsigned)iw < 28u) v = xb[ci * HW + ih * W + iw];
      vb.s[i] = v;
      if (++kw == 3) { kw = 0; if (++kh == 3) { kh = 0; ++ci; } }
    }
    *(uint4*)(Bs + nLoc * 40 + kHalf)     = vb.q4[0];
    *(uint4*)(Bs + nLoc * 40 + kHalf + 8) = vb.q4[1];
    __syncthreads();
    FragAB a0, a1, b0, b1;
    frag_a(As, wm, lane, a0);      frag_a(As, wm + 16, lane, a1);
    frag_b(Bs, wn, lane, b0);      frag_b(Bs, wn + 16, lane, b1);
    acc[0][0] = mma_bf16(a0, b0, acc[0][0]);
    acc[0][1] = mma_bf16(a0, b1, acc[0][1]);
    acc[1][0] = mma_bf16(a1, b0, acc[1][0]);
    acc[1][1] = mma_bf16(a1, b1, acc[1][1]);
    __syncthreads();
  }
  const float v1b = b1b[0], v2a = b2a[0];
#pragma unroll
  for (int fm = 0; fm < 2; ++fm)
#pragma unroll
    for (int fn = 0; fn < 2; ++fn) {
      int co0 = mBase + wm + fm * 16 + ((lane >> 4) << 3);
      int nn = nBase + wn + fn * 16 + (lane & 15);
      int b2 = nn / HW, r2 = nn - b2 * HW;
      AccU u; u.v = acc[fm][fn];
      size_t base = (size_t)b2 * CP * HW + r2;
#pragma unroll
      for (int r8 = 0; r8 < 8; ++r8)
        relu1[base + (size_t)(co0 + r8) * HW] = f2bf(fmaxf(u.f[r8] + v1b, 0.f) + v2a);
    }
}

// ================= GEMM2: 3x3 s2 p1 transposed conv (256->128) =================
__global__ __launch_bounds__(256) void k_gemm2(const unsigned short* __restrict__ relu1,
                                               const unsigned short* __restrict__ wp,
                                               float* __restrict__ mainBuf,
                                               const float* __restrict__ scl,
                                               const float* __restrict__ b2b) {
  __shared__ __align__(16) unsigned short As[64 * 40];
  __shared__ __align__(16) unsigned short Bs[128 * 40];
  const int tid = threadIdx.x, lane = tid & 31, wave = tid >> 5;
  const int wm = (wave & 1) << 5, wn = (wave >> 1) << 5;
  const int mBase = blockIdx.y << 6, nBase = blockIdx.x << 7;
  const unsigned AsOff = (unsigned)(uintptr_t)(&As[0]);

  const int nLoc = tid & 127;
  const int kHalf = (tid >> 7) << 4;
  const int n = nBase + nLoc;
  const bool nOK = n < N2;
  int bb = 0, oh = 0, ow = 0;
  if (nOK) { bb = n / 3025; int rp = n - bb * 3025; oh = rp / OH2; ow = rp - oh * OH2; }
  const int ohp = oh + 1, owp = ow + 1;
  const unsigned short* rb = relu1 + (size_t)bb * CP * HW;

  v8f acc[2][2] = {};
  for (int kc = 0; kc < KK; kc += 32) {
    if (tid == 0) {
      tdm_load_a(wp + (size_t)mBase * KK + kc, AsOff, KK);
      __builtin_amdgcn_s_wait_tensorcnt(0);
    }
    int kg0 = kc + kHalf;
    int ci = kg0 / 9, q = kg0 - ci * 9;
    int kh = q / 3,  kw = q - kh * 3;
    VBuf vb;
#pragma unroll
    for (int i = 0; i < 16; ++i) {
      int th = ohp - kh, tw = owp - kw;
      unsigned short v = 0;
      if (nOK && ((th | tw) >= 0) && !((th | tw) & 1)) {   // both >=0 and even
        int ih = th >> 1, iw = tw >> 1;
        if (ih < H && iw < W) v = rb[ci * HW + ih * W + iw];
      }
      vb.s[i] = v;
      if (++kw == 3) { kw = 0; if (++kh == 3) { kh = 0; ++ci; } }
    }
    *(uint4*)(Bs + nLoc * 40 + kHalf)     = vb.q4[0];
    *(uint4*)(Bs + nLoc * 40 + kHalf + 8) = vb.q4[1];
    __syncthreads();
    FragAB a0, a1, b0, b1;
    frag_a(As, wm, lane, a0);      frag_a(As, wm + 16, lane, a1);
    frag_b(Bs, wn, lane, b0);      frag_b(Bs, wn + 16, lane, b1);
    acc[0][0] = mma_bf16(a0, b0, acc[0][0]);
    acc[0][1] = mma_bf16(a0, b1, acc[0][1]);
    acc[1][0] = mma_bf16(a1, b0, acc[1][0]);
    acc[1][1] = mma_bf16(a1, b1, acc[1][1]);
    __syncthreads();
  }
  const float s = scl[0], bb2 = b2b[0];
#pragma unroll
  for (int fm = 0; fm < 2; ++fm)
#pragma unroll
    for (int fn = 0; fn < 2; ++fn) {
      int co0 = mBase + wm + fm * 16 + ((lane >> 4) << 3);
      int nn = nBase + wn + fn * 16 + (lane & 15);
      if (nn >= N2) continue;
      int b2 = nn / 3025, r2 = nn - b2 * 3025;
      AccU u; u.v = acc[fm][fn];
#pragma unroll
      for (int r8 = 0; r8 < 8; ++r8)
        mainBuf[(size_t)(b2 * CI + co0 + r8) * 3025 + r2] = u.f[r8] * s + bb2;
    }
}

// ======== GEMM3: 2x2 s2 up-branch (per-parity 1x1) fused with pad-add + relu ========
__global__ __launch_bounds__(256) void k_gemm3(const unsigned short* __restrict__ xsel,
                                               const unsigned short* __restrict__ wp3,
                                               const float* __restrict__ mainBuf,
                                               float* __restrict__ out) {
  __shared__ __align__(16) unsigned short As[64 * 40];
  __shared__ __align__(16) unsigned short Bs[128 * 40];
  const int tid = threadIdx.x, lane = tid & 31, wave = tid >> 5;
  const int wm = (wave & 1) << 5, wn = (wave >> 1) << 5;
  const int mBase = blockIdx.y << 6, nBase = blockIdx.x << 7;
  const int p = blockIdx.z, pi = p >> 1, pj = p & 1;
  const unsigned short* wp = wp3 + (size_t)p * CI * CI;
  const unsigned AsOff = (unsigned)(uintptr_t)(&As[0]);

  const int nLoc = tid & 127;
  const int kHalf = (tid >> 7) << 4;
  const int n = nBase + nLoc;
  const int bb = n / HW, rp = n - bb * HW;
  const unsigned short* xs = xsel + (size_t)bb * CI * HW + rp;

  v8f acc[2][2] = {};
  for (int kc = 0; kc < CI; kc += 32) {
    if (tid == 0) {
      tdm_load_a(wp + (size_t)mBase * CI + kc, AsOff, CI);
      __builtin_amdgcn_s_wait_tensorcnt(0);
    }
    VBuf vb;
#pragma unroll
    for (int i = 0; i < 16; ++i)
      vb.s[i] = xs[(size_t)(kc + kHalf + i) * HW];
    *(uint4*)(Bs + nLoc * 40 + kHalf)     = vb.q4[0];
    *(uint4*)(Bs + nLoc * 40 + kHalf + 8) = vb.q4[1];
    __syncthreads();
    FragAB a0, a1, b0, b1;
    frag_a(As, wm, lane, a0);      frag_a(As, wm + 16, lane, a1);
    frag_b(Bs, wn, lane, b0);      frag_b(Bs, wn + 16, lane, b1);
    acc[0][0] = mma_bf16(a0, b0, acc[0][0]);
    acc[0][1] = mma_bf16(a0, b1, acc[0][1]);
    acc[1][0] = mma_bf16(a1, b0, acc[1][0]);
    acc[1][1] = mma_bf16(a1, b1, acc[1][1]);
    __syncthreads();
  }
#pragma unroll
  for (int fm = 0; fm < 2; ++fm)
#pragma unroll
    for (int fn = 0; fn < 2; ++fn) {
      int co0 = mBase + wm + fm * 16 + ((lane >> 4) << 3);
      int nn = nBase + wn + fn * 16 + (lane & 15);
      int b2 = nn / HW, r2 = nn - b2 * HW;
      int i2 = r2 / W, j2 = r2 - i2 * W;
      int oi = (i2 << 1) + pi, oj = (j2 << 1) + pj;
      AccU u; u.v = acc[fm][fn];
#pragma unroll
      for (int r8 = 0; r8 < 8; ++r8) {
        int co = co0 + r8;
        float mv = 0.f;
        if (oi > 0 && oj > 0)
          mv = mainBuf[(size_t)(b2 * CI + co) * (OH2 * OH2) + (oi - 1) * OH2 + (oj - 1)];
        out[(size_t)(b2 * CI + co) * 3136 + oi * 56 + oj] = fmaxf(mv + u.f[r8], 0.f);
      }
    }
}

// ================= launch =================
extern "C" void kernel_launch(void* const* d_in, const int* in_sizes, int n_in,
                              void* d_out, int out_size, void* d_ws, size_t ws_size,
                              hipStream_t stream) {
  (void)in_sizes; (void)n_in; (void)out_size; (void)ws_size;
  const float* x   = (const float*)d_in[0];
  const float* w2  = (const float*)d_in[1];   // w_conv2 [256,256,3,3]
  const float* w1  = (const float*)d_in[2];   // w_conv1 [256,128,3,3]
  const float* wu  = (const float*)d_in[3];   // w_up    [128,128,2,2]
  const float* b1a = (const float*)d_in[4];
  const float* b1b = (const float*)d_in[5];
  const float* b2a = (const float*)d_in[6];
  const float* b2b = (const float*)d_in[7];
  const float* scl = (const float*)d_in[8];
  float* out = (float*)d_out;

  char* ws = (char*)d_ws;
  unsigned short* xbf   = (unsigned short*)(ws + OFF_XBF);
  unsigned short* relu1 = (unsigned short*)(ws + OFF_RELU);
  float*          mainB = (float*)(ws + OFF_MAIN);
  unsigned short* xsel  = (unsigned short*)(ws + OFF_XSEL);
  unsigned short* wp1   = (unsigned short*)(ws + OFF_W1);
  unsigned short* wp2   = (unsigned short*)(ws + OFF_W2);
  unsigned short* wp3   = (unsigned short*)(ws + OFF_W3);
  float*          rng   = (float*)(ws + OFF_RNG);
  int*            idx   = (int*)(ws + OFF_IDX);

  const int nX = BN * CP * HW;
  hipLaunchKernelGGL(k_pack_x,  dim3((nX + 255) / 256), dim3(256), 0, stream, x, b1a, xbf, nX);
  hipLaunchKernelGGL(k_pack_w2, dim3((CP * KK + 255) / 256), dim3(256), 0, stream, w2, wp1);
  hipLaunchKernelGGL(k_pack_w1, dim3((CI * KK + 255) / 256), dim3(256), 0, stream, w1, wp2);
  hipLaunchKernelGGL(k_pack_wup,dim3((4 * CI * CI + 255) / 256), dim3(256), 0, stream, wu, wp3);

  hipLaunchKernelGGL(k_range,   dim3(BN * CP), dim3(128), 0, stream, x, rng);
  hipLaunchKernelGGL(k_argsort, dim3(BN), dim3(256), 0, stream, rng, idx);
  hipLaunchKernelGGL(k_gather,  dim3((BN * CI * HW + 255) / 256), dim3(256), 0, stream, xbf, idx, xsel);

  hipLaunchKernelGGL(k_gemm1, dim3(N1 / 128, CP / 64), dim3(256), 0, stream,
                     xbf, wp1, relu1, b1b, b2a);
  hipLaunchKernelGGL(k_gemm2, dim3((N2 + 127) / 128, CI / 64), dim3(256), 0, stream,
                     relu1, wp2, mainB, scl, b2b);
  hipLaunchKernelGGL(k_gemm3, dim3(N3 / 128, CI / 64, 4), dim3(256), 0, stream,
                     xsel, wp3, mainB, out);
}
